// VectorQuantizer_5652176961803
// MI455X (gfx1250) — matrix operand order repaired and checked by reading, not compiled, not run on hardware
//
#include <hip/hip_runtime.h>
#include <hip/hip_bf16.h>

// ---- CDNA5 WMMA types ------------------------------------------------------
typedef __attribute__((ext_vector_type(16))) __bf16       v16bf;
typedef __attribute__((ext_vector_type(8)))  float        v8f;
typedef __attribute__((ext_vector_type(4)))  unsigned int u32x4;

union BF16x16 { v16bf v; u32x4 q[2]; };

#define DIM    256
#define NROWS  8192      // B*H*W = 8*32*32
#define NCODES 16384
#define NELEMS 2097152   // 8*256*32*32

// ---------------------------------------------------------------------------
// Kernel 1: normalize z rows (transpose b,c,h,w -> [row=bhw, c]) to bf16.
// ---------------------------------------------------------------------------
__global__ __launch_bounds__(256) void vq_normalize_z_kernel(
    const float* __restrict__ z, __bf16* __restrict__ A)
{
    __shared__ float tile[256][33];           // [c][hw_local], pad kills conflicts
    const int b      = blockIdx.x >> 5;
    const int hwbase = (blockIdx.x & 31) << 5;
    const int lane   = threadIdx.x & 31;
    const int wave   = threadIdx.x >> 5;

    for (int it = 0; it < 32; ++it) {
        const int c = it * 8 + wave;
        tile[c][lane] = z[((size_t)(b * 256 + c) << 10) + hwbase + lane];
    }
    __syncthreads();

    for (int rr = 0; rr < 4; ++rr) {
        const int r = wave * 4 + rr;
        float s = 0.0f;
        #pragma unroll
        for (int j = 0; j < 8; ++j) {
            const float x = tile[lane + 32 * j][r];
            s += x * x;
        }
        #pragma unroll
        for (int m = 16; m >= 1; m >>= 1) s += __shfl_xor(s, m, 32);
        const float sc = 1.0f / fmaxf(sqrtf(s), 1e-12f);

        const size_t i = ((size_t)b << 10) + hwbase + r;
        #pragma unroll
        for (int j = 0; j < 8; ++j) {
            const int c = lane + 32 * j;
            A[i * DIM + c] = (__bf16)(tile[c][r] * sc);
        }
    }
}

// ---------------------------------------------------------------------------
// Kernel 2: normalize codebook rows to bf16 (rows already contiguous).
// ---------------------------------------------------------------------------
__global__ __launch_bounds__(256) void vq_normalize_cb_kernel(
    const float* __restrict__ cb, __bf16* __restrict__ Bm)
{
    const int lane = threadIdx.x & 31;
    const int wave = threadIdx.x >> 5;
    const int row  = blockIdx.x * 8 + wave;
    const float* src = cb + (size_t)row * DIM;

    float vals[8];
    float s = 0.0f;
    #pragma unroll
    for (int j = 0; j < 8; ++j) {
        vals[j] = src[lane + 32 * j];
        s += vals[j] * vals[j];
    }
    #pragma unroll
    for (int m = 16; m >= 1; m >>= 1) s += __shfl_xor(s, m, 32);
    const float sc = 1.0f / fmaxf(sqrtf(s), 1e-12f);

    __bf16* dst = Bm + (size_t)row * DIM;
    #pragma unroll
    for (int j = 0; j < 8; ++j) dst[lane + 32 * j] = (__bf16)(vals[j] * sc);
}

// ---------------------------------------------------------------------------
// Kernel 3: fused bf16 WMMA GEMM + row-argmax.
// 4-deep circular LDS pipeline fed by GLOBAL_LOAD_ASYNC_TO_LDS_B128, 3 tiles
// in flight (s_wait_asynccnt 0x8 => oldest tile landed, in-order async).
// 32-column tiles: two independent 8-WMMA accumulation chains per iteration
// (c0/c1), halving barrier count and doubling the copy-landing window.
// grid = (64 M-blocks, 4 N-splits), 256 threads = 8 waves.
// ---------------------------------------------------------------------------
__global__ __launch_bounds__(256) void vq_argmax_kernel(
    const __bf16* __restrict__ A, const __bf16* __restrict__ Bm,
    unsigned long long* __restrict__ keys)
{
    __shared__ __bf16 bt[4][32][264];         // 4 buffers, 32 rows, 528B stride
    const int tid   = threadIdx.x;
    const int lane  = tid & 31;
    const int wave  = tid >> 5;
    const int Mbase = blockIdx.x * 128 + wave * 16;
    const int ns    = blockIdx.y;

    // A fragments per ISA 16-bit A layout: lane<16 -> M=lane, K in {0..7,16..23};
    // lane>=16 -> M=lane-16, K in {8..15,24..31} (per 32-wide K step).
    v16bf afrag[8];
    {
        const int row   = Mbase + (lane & 15);
        const int khalf = (lane >> 4) << 3;   // 0 or 8
        const u32x4* Arow = reinterpret_cast<const u32x4*>(A + (size_t)row * DIM);
        #pragma unroll
        for (int kk = 0; kk < 8; ++kk) {
            BF16x16 f;
            f.q[0] = Arow[(kk * 32 + khalf) >> 3];
            f.q[1] = Arow[(kk * 32 + 16 + khalf) >> 3];
            afrag[kk] = f.v;
        }
    }

    float best[8];
    int   bestn[8];
    #pragma unroll
    for (int v = 0; v < 8; ++v) { best[v] = -3.0e38f; bestn[v] = 0; }

    // B fragment addressing: lane -> N = lane%16, K in {8h..8h+7, 16+8h..16+8h+7}, h = lane/16
    const int n0     = lane & 15;
    const int khalfB = (lane >> 4) << 3;

    // staging: 32 rows x 512B per tile; 64B per thread (4 async b128)
    const int r   = tid >> 3;                 // 0..31 tile row
    const int seg = tid & 7;                  // 0..7, 64B segment
    const unsigned ldsbase =                   // flat-LDS low 32 bits == offset
        (unsigned)(uintptr_t)(&bt[0][0][0]) + (unsigned)(r * 528 + seg * 64);
    #define BUFBYTES (32u * 264u * 2u)         // 16896

    // prologue: issue tiles 0..2 into buffers 0..2 (12 async ops/wave)
    #pragma unroll
    for (int p = 0; p < 3; ++p) {
        const unsigned dsto = ldsbase + (unsigned)p * BUFBYTES;
        const __bf16*  g    = Bm + (size_t)(ns * 4096 + p * 32 + r) * DIM + seg * 32;
        asm volatile(
            "global_load_async_to_lds_b128 %0, %1, off\n\t"
            "global_load_async_to_lds_b128 %0, %1, off offset:16\n\t"
            "global_load_async_to_lds_b128 %0, %1, off offset:32\n\t"
            "global_load_async_to_lds_b128 %0, %1, off offset:48"
            :: "v"(dsto), "v"(g) : "memory");
    }

    for (int t = 0; t < 128; ++t) {
        // oldest in-flight tile (t) has landed for this wave; publish block-wide
        asm volatile("s_wait_asynccnt 0x8" ::: "memory");
        __syncthreads();

        // issue tile t+3 into buffer (t+3)&3; its last readers (tile t-1)
        // are ordered behind the barrier above. Index wraps in the tail so
        // the wait immediate stays constant (redundant copies are harmless).
        {
            const int tn = (t + 3) & 127;
            const unsigned dsto = ldsbase + (unsigned)((t + 3) & 3) * BUFBYTES;
            const __bf16*  g    = Bm + (size_t)(ns * 4096 + tn * 32 + r) * DIM + seg * 32;
            asm volatile(
                "global_load_async_to_lds_b128 %0, %1, off\n\t"
                "global_load_async_to_lds_b128 %0, %1, off offset:16\n\t"
                "global_load_async_to_lds_b128 %0, %1, off offset:32\n\t"
                "global_load_async_to_lds_b128 %0, %1, off offset:48"
                :: "v"(dsto), "v"(g) : "memory");
        }

        const int buf = t & 3;
        v8f c0 = {}, c1 = {};
        #pragma unroll
        for (int kk = 0; kk < 8; ++kk) {
            BF16x16 b0, b1;
            const u32x4* l0 =
                reinterpret_cast<const u32x4*>(&bt[buf][n0][kk * 32 + khalfB]);
            const u32x4* l1 =
                reinterpret_cast<const u32x4*>(&bt[buf][16 + n0][kk * 32 + khalfB]);
            b0.q[0] = l0[0]; b0.q[1] = l0[2];
            b1.q[0] = l1[0]; b1.q[1] = l1[2];
            c0 = __builtin_amdgcn_wmma_f32_16x16x32_bf16(
                     false, afrag[kk], false, b0.v, (short)0, c0, false, false);
            c1 = __builtin_amdgcn_wmma_f32_16x16x32_bf16(
                     false, afrag[kk], false, b1.v, (short)0, c1, false, false);
        }

        // C layout: N = lane%16, M = vgpr + 8*(lane/16)
        const int na = ns * 4096 + t * 32 + n0;
        const int nb = na + 16;
        #pragma unroll
        for (int v = 0; v < 8; ++v)
            if (c0[v] > best[v]) { best[v] = c0[v]; bestn[v] = na; }
        #pragma unroll
        for (int v = 0; v < 8; ++v)
            if (c1[v] > best[v]) { best[v] = c1[v]; bestn[v] = nb; }
    }

    // reduce across the 16 lanes sharing each M row (xor masks stay in-group)
    #pragma unroll
    for (int v = 0; v < 8; ++v) {
        float bv = best[v];
        int   bn = bestn[v];
        #pragma unroll
        for (int m = 8; m >= 1; m >>= 1) {
            const float ov = __shfl_xor(bv, m, 32);
            const int   on = __shfl_xor(bn, m, 32);
            if (ov > bv || (ov == bv && on < bn)) { bv = ov; bn = on; }
        }
        if ((lane & 15) == 0) {
            const int M = Mbase + v + ((lane >> 4) << 3);
            const unsigned int u   = __float_as_uint(bv);
            const unsigned int ord = (u & 0x80000000u) ? ~u : (u | 0x80000000u);
            const unsigned long long key =
                ((unsigned long long)ord << 32) | (unsigned long long)(16383 - bn);
            atomicMax(&keys[M], key);
        }
    }
}

// ---------------------------------------------------------------------------
// Kernel 4: unpack argmax keys -> int index (ws) + float index (d_out tail).
// ---------------------------------------------------------------------------
__global__ __launch_bounds__(256) void vq_extract_idx_kernel(
    const unsigned long long* __restrict__ keys,
    int* __restrict__ idxw, float* __restrict__ outIdx)
{
    const int i = blockIdx.x * 256 + threadIdx.x;
    const unsigned long long k = keys[i];
    const int id = 16383 - (int)(unsigned int)(k & 0xFFFFFFFFull);
    idxw[i]   = id;
    outIdx[i] = (float)id;
}

// ---------------------------------------------------------------------------
// Kernel 5: gather raw codebook rows back to (b,c,h,w) and accumulate
// loss = 1.25 * mean((z_q - zt)^2).
// ---------------------------------------------------------------------------
__global__ __launch_bounds__(256) void vq_gather_loss_kernel(
    const float* __restrict__ z, const float* __restrict__ cb,
    const int* __restrict__ idxw, float* __restrict__ out)
{
    const unsigned int t = blockIdx.x * 256 + threadIdx.x;   // < NELEMS
    const int b  = t >> 18;
    const int c  = (t >> 10) & 255;
    const int hw = t & 1023;
    const int i  = (b << 10) | hw;                            // flat bhw row
    const int id = idxw[i];

    const float q  = cb[(size_t)id * DIM + c];                // raw codebook gather
    const float zt = z[t];
    out[1 + t] = q;                                           // z_q_out == z_q by value

    const float d = q - zt;
    float s = d * d;
    #pragma unroll
    for (int m = 16; m >= 1; m >>= 1) s += __shfl_xor(s, m, 32);

    __shared__ float ps[8];
    const int lane = threadIdx.x & 31;
    const int wave = threadIdx.x >> 5;
    if (lane == 0) ps[wave] = s;
    __syncthreads();
    if (wave == 0) {
        float bs = (lane < 8) ? ps[lane] : 0.0f;
        #pragma unroll
        for (int m = 4; m >= 1; m >>= 1) bs += __shfl_xor(bs, m, 32);
        if (lane == 0) atomicAdd(&out[0], bs * (1.25f / (float)NELEMS));
    }
}

// ---------------------------------------------------------------------------
extern "C" void kernel_launch(void* const* d_in, const int* in_sizes, int n_in,
                              void* d_out, int out_size, void* d_ws, size_t ws_size,
                              hipStream_t stream)
{
    const float* z  = (const float*)d_in[0];   // (8,256,32,32) f32
    const float* cb = (const float*)d_in[1];   // (16384,256)   f32
    float* out = (float*)d_out;                // [loss | z_q_out (2M) | indices (8192)]
    char*  ws  = (char*)d_ws;

    __bf16* A  = (__bf16*)ws;                                         // 4 MiB
    __bf16* Bm = (__bf16*)(ws + (size_t)4 * 1024 * 1024);             // 8 MiB
    unsigned long long* keys =
        (unsigned long long*)(ws + (size_t)12 * 1024 * 1024);         // 64 KiB
    int* idxw = (int*)(ws + (size_t)12 * 1024 * 1024 + 65536);        // 32 KiB

    hipMemsetAsync(keys, 0, NROWS * sizeof(unsigned long long), stream);
    hipMemsetAsync(out, 0, sizeof(float), stream);                    // loss accumulator

    vq_normalize_z_kernel <<<256,  256, 0, stream>>>(z, A);
    vq_normalize_cb_kernel<<<2048, 256, 0, stream>>>(cb, Bm);
    vq_argmax_kernel      <<<dim3(64, 4), 256, 0, stream>>>(A, Bm, keys);
    vq_extract_idx_kernel <<<32,   256, 0, stream>>>(keys, idxw, out + 1 + NELEMS);
    vq_gather_loss_kernel <<<8192, 256, 0, stream>>>(z, cb, idxw, out);
}
